// MultiHeadAttention_67577015435575
// MI455X (gfx1250) — matrix operand order repaired and checked by reading, not compiled, not run on hardware
//
#include <hip/hip_runtime.h>

// ---------------------------------------------------------------------------
// CDNA5 (gfx1250) WMMA multi-head attention forward
//   out  = (softmax(QK^T/sqrt(64) + symmask) V) Wo + bo
//   attn = softmax(...)                       (both fp32 outputs, concatenated)
// All matmuls via v_wmma_f32_16x16x32_bf16 (bf16 in, f32 accumulate).
// ---------------------------------------------------------------------------

constexpr int BN    = 2;
constexpr int SN    = 2048;
constexpr int DN    = 1024;
constexpr int HN    = 16;
constexpr int DEPTH = 64;
constexpr int MROWS = BN * SN;  // 4096

typedef __attribute__((ext_vector_type(16))) __bf16 v16bf;
typedef __attribute__((ext_vector_type(8)))  float  v8f;

union FragBF {
  v16bf v;
  uint4 q[2];
  unsigned short s[16];
};

__device__ __forceinline__ unsigned short f2bf(float f) {
  unsigned int u = __builtin_bit_cast(unsigned int, f);
  u += 0x7FFFu + ((u >> 16) & 1u);                 // round-to-nearest-even
  return (unsigned short)(u >> 16);
}

__device__ __forceinline__ v8f wmma_bf16(const FragBF& a, const FragBF& b, v8f c) {
  return __builtin_amdgcn_wmma_f32_16x16x32_bf16(false, a.v, false, b.v,
                                                 (short)0, c, false, false);
}

// ---------------------------------------------------------------------------
// elementwise fp32 -> bf16
__global__ void cvt_bf16_kernel(const float* __restrict__ src,
                                unsigned short* __restrict__ dst, int n) {
  int i = blockIdx.x * blockDim.x + threadIdx.x;
  if (i < n) dst[i] = f2bf(src[i]);
}

// W[K][N] fp32 -> Wt[N][K] bf16   (K = N = DN)
__global__ void transpose_bf16_kernel(const float* __restrict__ W,
                                      unsigned short* __restrict__ Wt) {
  int i = blockIdx.x * blockDim.x + threadIdx.x;
  int nn = i / DN;
  int k  = i % DN;
  Wt[i] = f2bf(W[(size_t)k * DN + nn]);
}

// ---------------------------------------------------------------------------
// C = A(bf16, MROWSxDN row-major) * Bt(bf16, DNxDN row-major == B^T) + bias
// 256 threads = 8 waves; each wave computes a 16x64 C tile; WG = 32(M)x256(N).
// Software-pipelined: fragments for k+32 load while k's 4 WMMAs execute.
// MODE 0: fp32 C row-major       MODE 1: bf16 C row-major
// MODE 2: bf16 Vt layout: Vt[(row/SN)*DN + col][row%SN]  (per-head V transpose)
template <int MODE>
__global__ __launch_bounds__(256) void gemm_bf16_kernel(
    const unsigned short* __restrict__ A,
    const unsigned short* __restrict__ Bt,
    const float* __restrict__ bias,
    void* __restrict__ Cout) {
  constexpr int K = DN;  // 1024
  constexpr int N = DN;  // 1024
  int lane = threadIdx.x & 31;
  int w    = threadIdx.x >> 5;
  int m0   = blockIdx.y * 32  + (w >> 2) * 16;
  int n0   = blockIdx.x * 256 + (w & 3)  * 64;
  int ln   = lane & 15;   // A: row m ; B/C: col n
  int lg   = lane >> 4;   // half-wave selector

  v8f c[4];
#pragma unroll
  for (int t = 0; t < 4; ++t)
#pragma unroll
    for (int v = 0; v < 8; ++v) c[t][v] = 0.0f;

  const unsigned short* arow = A + (size_t)(m0 + ln) * K + 16 * lg;
  const unsigned short* br0  = Bt + (size_t)(n0 + 0 * 16 + ln) * K + 16 * lg;
  const unsigned short* br1  = Bt + (size_t)(n0 + 1 * 16 + ln) * K + 16 * lg;
  const unsigned short* br2  = Bt + (size_t)(n0 + 2 * 16 + ln) * K + 16 * lg;
  const unsigned short* br3  = Bt + (size_t)(n0 + 3 * 16 + ln) * K + 16 * lg;

#define LOAD_A(f, kk)                                  \
  {                                                    \
    (f).q[0] = *(const uint4*)(arow + (kk));           \
    (f).q[1] = *(const uint4*)(arow + (kk) + 8);       \
  }
#define LOAD_B(f, base, kk)                            \
  {                                                    \
    (f).q[0] = *(const uint4*)((base) + (kk));         \
    (f).q[1] = *(const uint4*)((base) + (kk) + 8);     \
  }

  FragBF fa, fb0, fb1, fb2, fb3;
  LOAD_A(fa, 0);
  LOAD_B(fb0, br0, 0);
  LOAD_B(fb1, br1, 0);
  LOAD_B(fb2, br2, 0);
  LOAD_B(fb3, br3, 0);

#pragma unroll 4
  for (int k0 = 0; k0 < K - 32; k0 += 32) {
    FragBF na, nb0, nb1, nb2, nb3;
    LOAD_A(na, k0 + 32);
    LOAD_B(nb0, br0, k0 + 32);
    LOAD_B(nb1, br1, k0 + 32);
    LOAD_B(nb2, br2, k0 + 32);
    LOAD_B(nb3, br3, k0 + 32);
    __builtin_prefetch(arow + k0 + 96, 0, 0);
    __builtin_prefetch(br0 + k0 + 96, 0, 0);
    __builtin_prefetch(br1 + k0 + 96, 0, 0);
    __builtin_prefetch(br2 + k0 + 96, 0, 0);
    __builtin_prefetch(br3 + k0 + 96, 0, 0);
    c[0] = wmma_bf16(fa, fb0, c[0]);
    c[1] = wmma_bf16(fa, fb1, c[1]);
    c[2] = wmma_bf16(fa, fb2, c[2]);
    c[3] = wmma_bf16(fa, fb3, c[3]);
    fa = na; fb0 = nb0; fb1 = nb1; fb2 = nb2; fb3 = nb3;
  }
  c[0] = wmma_bf16(fa, fb0, c[0]);
  c[1] = wmma_bf16(fa, fb1, c[1]);
  c[2] = wmma_bf16(fa, fb2, c[2]);
  c[3] = wmma_bf16(fa, fb3, c[3]);
#undef LOAD_A
#undef LOAD_B

#pragma unroll
  for (int t = 0; t < 4; ++t) {
    int col = n0 + t * 16 + ln;
    float bv = bias[col];
#pragma unroll
    for (int v = 0; v < 8; ++v) {
      int   row = m0 + v + 8 * lg;
      float val = c[t][v] + bv;
      if constexpr (MODE == 0) {
        ((float*)Cout)[(size_t)row * N + col] = val;
      } else if constexpr (MODE == 1) {
        ((unsigned short*)Cout)[(size_t)row * N + col] = f2bf(val);
      } else {  // per-head transposed V: Vt[b*DN + col][s]
        ((unsigned short*)Cout)[((size_t)((row / SN) * N + col)) * SN +
                                (row % SN)] = f2bf(val);
      }
    }
  }
}

// ---------------------------------------------------------------------------
// Attention: 4 waves (128 threads) per (b, h, 16-query-row block), sharing one
// 16 x SN fp32 logits tile (128 KB) in LDS. attn written to HBM exactly once.
__global__ __launch_bounds__(128) void attn_kernel(
    const unsigned short* __restrict__ Qbf,   // (B*S, D) bf16
    const unsigned short* __restrict__ Kbf,   // (B*S, D) bf16
    const unsigned short* __restrict__ Vt,    // (B*H*64, S) bf16
    const float* __restrict__ mask,           // (B, S)
    float* __restrict__ attnOut,              // (B, H, S, S)
    unsigned short* __restrict__ zbf) {       // (B*S, D) bf16
  __shared__ float pbuf[16 * SN];     // 128 KB logits/probs tile
  __shared__ float zred[4][16 * 64];  // 16 KB partial-z reduction
  __shared__ float srowmax[16];
  __shared__ float srowsum[16];
  __shared__ float smq[16];

  int tid  = threadIdx.x;
  int w    = tid >> 5;
  int lane = tid & 31;
  int qb = blockIdx.x % (SN / 16);
  int bh = blockIdx.x / (SN / 16);
  int b  = bh / HN;
  int h  = bh % HN;
  int q0 = qb * 16;
  int ln = lane & 15;
  int lg = lane >> 4;

  if (tid < 16) smq[tid] = mask[b * SN + q0 + tid];
  __syncthreads();

  // loop-invariant Q fragments (K = 0..31 and 32..63 of this head)
  const unsigned short* qrow =
      Qbf + (size_t)(b * SN + q0 + ln) * DN + h * DEPTH + 16 * lg;
  FragBF fa0, fa1;
  fa0.q[0] = *(const uint4*)(qrow);
  fa0.q[1] = *(const uint4*)(qrow + 8);
  fa1.q[0] = *(const uint4*)(qrow + 32);
  fa1.q[1] = *(const uint4*)(qrow + 40);

  const float scale = 0.125f;  // 1/sqrt(64)

  // ---- pass 1: logits = QK^T * scale + min(1, mq+mk)  -> LDS
  //      waves interleave over 16-wide k blocks (stride 64 rows)
  {
    const unsigned short* krow =
        Kbf + (size_t)(b * SN + w * 16 + ln) * DN + h * DEPTH + 16 * lg;
    const float* mrow = mask + b * SN + w * 16 + ln;
    for (int kb = w * 16; kb < SN; kb += 64) {
      FragBF fb0, fb1;
      fb0.q[0] = *(const uint4*)(krow);
      fb0.q[1] = *(const uint4*)(krow + 8);
      fb1.q[0] = *(const uint4*)(krow + 32);
      fb1.q[1] = *(const uint4*)(krow + 40);
      v8f cc;
#pragma unroll
      for (int v = 0; v < 8; ++v) cc[v] = 0.0f;
      cc = wmma_bf16(fa0, fb0, cc);
      cc = wmma_bf16(fa1, fb1, cc);
      float mk = *mrow;
#pragma unroll
      for (int v = 0; v < 8; ++v) {
        int r = v + 8 * lg;
        pbuf[r * SN + kb + ln] = cc[v] * scale + fminf(1.0f, smq[r] + mk);
      }
      krow += (size_t)64 * DN;
      mrow += 64;
    }
  }
  __syncthreads();

  // ---- pass 2: row max (4 rows per wave, shuffle butterfly)
  for (int r = w * 4; r < w * 4 + 4; ++r) {
    float mx = -3.402823466e38f;
    for (int i = lane; i < SN; i += 32) mx = fmaxf(mx, pbuf[r * SN + i]);
#pragma unroll
    for (int off = 16; off; off >>= 1) mx = fmaxf(mx, __shfl_xor(mx, off, 32));
    if (lane == 0) srowmax[r] = mx;
  }
  __syncthreads();

  // ---- pass 3: exp in place + row sum (4 rows per wave)
  for (int r = w * 4; r < w * 4 + 4; ++r) {
    float mxr = srowmax[r];
    float s = 0.0f;
    for (int i = lane; i < SN; i += 32) {
      float p = __expf(pbuf[r * SN + i] - mxr);
      pbuf[r * SN + i] = p;
      s += p;
    }
#pragma unroll
    for (int off = 16; off; off >>= 1) s += __shfl_xor(s, off, 32);
    if (lane == 0) srowsum[r] = s;
  }
  __syncthreads();

  // ---- pass 4: partial z = p @ V over this wave's quarter of S
  v8f z[4];
#pragma unroll
  for (int t = 0; t < 4; ++t)
#pragma unroll
    for (int v = 0; v < 8; ++v) z[t][v] = 0.0f;
  {
    int kb0 = w * (SN / 4);
    const float* prow = pbuf + ln * SN + kb0 + 16 * lg;
    const unsigned short* vtb =
        Vt + (size_t)(bh * 64) * SN + kb0 + 16 * lg;
    const unsigned short* vr0 = vtb + (size_t)(0 * 16 + ln) * SN;
    const unsigned short* vr1 = vtb + (size_t)(1 * 16 + ln) * SN;
    const unsigned short* vr2 = vtb + (size_t)(2 * 16 + ln) * SN;
    const unsigned short* vr3 = vtb + (size_t)(3 * 16 + ln) * SN;
    for (int kb = 0; kb < SN / 4; kb += 32) {
      float4 pa = *(const float4*)(prow + kb);
      float4 pb = *(const float4*)(prow + kb + 4);
      float4 pc = *(const float4*)(prow + kb + 8);
      float4 pd = *(const float4*)(prow + kb + 12);
      FragBF fa;
      fa.s[0]  = f2bf(pa.x); fa.s[1]  = f2bf(pa.y);
      fa.s[2]  = f2bf(pa.z); fa.s[3]  = f2bf(pa.w);
      fa.s[4]  = f2bf(pb.x); fa.s[5]  = f2bf(pb.y);
      fa.s[6]  = f2bf(pb.z); fa.s[7]  = f2bf(pb.w);
      fa.s[8]  = f2bf(pc.x); fa.s[9]  = f2bf(pc.y);
      fa.s[10] = f2bf(pc.z); fa.s[11] = f2bf(pc.w);
      fa.s[12] = f2bf(pd.x); fa.s[13] = f2bf(pd.y);
      fa.s[14] = f2bf(pd.z); fa.s[15] = f2bf(pd.w);
      FragBF fb0, fb1, fb2, fb3;
      fb0.q[0] = *(const uint4*)(vr0 + kb);
      fb0.q[1] = *(const uint4*)(vr0 + kb + 8);
      fb1.q[0] = *(const uint4*)(vr1 + kb);
      fb1.q[1] = *(const uint4*)(vr1 + kb + 8);
      fb2.q[0] = *(const uint4*)(vr2 + kb);
      fb2.q[1] = *(const uint4*)(vr2 + kb + 8);
      fb3.q[0] = *(const uint4*)(vr3 + kb);
      fb3.q[1] = *(const uint4*)(vr3 + kb + 8);
      z[0] = wmma_bf16(fa, fb0, z[0]);
      z[1] = wmma_bf16(fa, fb1, z[1]);
      z[2] = wmma_bf16(fa, fb2, z[2]);
      z[3] = wmma_bf16(fa, fb3, z[3]);
    }
  }
  // stash this wave's partial 16x64 tile
#pragma unroll
  for (int t = 0; t < 4; ++t)
#pragma unroll
    for (int v = 0; v < 8; ++v)
      zred[w][(v + 8 * lg) * 64 + t * 16 + ln] = z[t][v];

  // ---- pass 5: write normalized attn once (4 rows per wave, float4 stores)
  float* aout = attnOut + ((size_t)bh * SN + q0) * SN;
  for (int r = w * 4; r < w * 4 + 4; ++r) {
    float inv = 1.0f / srowsum[r];
    for (int i = lane; i < SN / 4; i += 32) {
      float4 p = *(const float4*)(pbuf + r * SN + i * 4);
      p.x *= inv; p.y *= inv; p.z *= inv; p.w *= inv;
      *(float4*)(aout + (size_t)r * SN + i * 4) = p;
    }
  }
  __syncthreads();

  // ---- pass 6: reduce partial z across waves, normalize, store bf16
  for (int idx = tid; idx < 16 * 64; idx += 128) {
    float s = zred[0][idx] + zred[1][idx] + zred[2][idx] + zred[3][idx];
    int r = idx >> 6;
    int d = idx & 63;
    zbf[(size_t)(b * SN + q0 + r) * DN + h * DEPTH + d] =
        f2bf(s / srowsum[r]);
  }
}

// ---------------------------------------------------------------------------
extern "C" void kernel_launch(void* const* d_in, const int* in_sizes, int n_in,
                              void* d_out, int out_size, void* d_ws,
                              size_t ws_size, hipStream_t stream) {
  (void)in_sizes; (void)n_in; (void)out_size; (void)ws_size;
  const float* x    = (const float*)d_in[0];
  const float* mask = (const float*)d_in[1];
  const float* Wq   = (const float*)d_in[2];
  const float* bq   = (const float*)d_in[3];
  const float* Wk   = (const float*)d_in[4];
  const float* bk   = (const float*)d_in[5];
  const float* Wv   = (const float*)d_in[6];
  const float* bv   = (const float*)d_in[7];
  const float* Wo   = (const float*)d_in[8];
  const float* bo   = (const float*)d_in[9];

  char* ws = (char*)d_ws;
  size_t off = 0;
  auto alloc = [&](size_t bytes) -> void* {
    void* p = ws + off;
    off += (bytes + 255) & ~(size_t)255;
    return p;
  };
  unsigned short* xbf = (unsigned short*)alloc((size_t)MROWS * DN * 2);
  unsigned short* WqT = (unsigned short*)alloc((size_t)DN * DN * 2);
  unsigned short* WkT = (unsigned short*)alloc((size_t)DN * DN * 2);
  unsigned short* WvT = (unsigned short*)alloc((size_t)DN * DN * 2);
  unsigned short* WoT = (unsigned short*)alloc((size_t)DN * DN * 2);
  unsigned short* Qbf = (unsigned short*)alloc((size_t)MROWS * DN * 2);
  unsigned short* Kbf = (unsigned short*)alloc((size_t)MROWS * DN * 2);
  unsigned short* Vt  = (unsigned short*)alloc((size_t)BN * HN * DEPTH * SN * 2);
  unsigned short* zbf = (unsigned short*)alloc((size_t)MROWS * DN * 2);

  // 1) precision conversion + weight transposes
  cvt_bf16_kernel<<<(MROWS * DN) / 256, 256, 0, stream>>>(x, xbf, MROWS * DN);
  transpose_bf16_kernel<<<(DN * DN) / 256, 256, 0, stream>>>(Wq, WqT);
  transpose_bf16_kernel<<<(DN * DN) / 256, 256, 0, stream>>>(Wk, WkT);
  transpose_bf16_kernel<<<(DN * DN) / 256, 256, 0, stream>>>(Wv, WvT);
  transpose_bf16_kernel<<<(DN * DN) / 256, 256, 0, stream>>>(Wo, WoT);

  // 2) Q/K/V projections (WMMA)
  dim3 gg(DN / 256, MROWS / 32);
  gemm_bf16_kernel<1><<<gg, 256, 0, stream>>>(xbf, WqT, bq, Qbf);
  gemm_bf16_kernel<1><<<gg, 256, 0, stream>>>(xbf, WkT, bk, Kbf);
  gemm_bf16_kernel<2><<<gg, 256, 0, stream>>>(xbf, WvT, bv, Vt);

  // 3) fused attention (logits + masked softmax in LDS + attn write + z)
  float* outp  = (float*)d_out;
  float* attnp = outp + (size_t)MROWS * DN;
  attn_kernel<<<BN * HN * (SN / 16), 128, 0, stream>>>(Qbf, Kbf, Vt, mask,
                                                       attnp, zbf);

  // 4) output projection (WMMA, fp32 out)
  gemm_bf16_kernel<0><<<gg, 256, 0, stream>>>(zbf, WoT, bo, outp);
}
